// GPRGNN_4337916969349
// MI455X (gfx1250) — matrix operand order repaired and checked by reading, not compiled, not run on hardware
//
#include <hip/hip_runtime.h>

#define N_NODES 100000
#define E_EDGES 1600000
#define IN_CH   500
#define HID     256
#define OUT_CH  40
#define K_HOPS  10

typedef __attribute__((ext_vector_type(16))) _Float16 v16h;
typedef __attribute__((ext_vector_type(8)))  float    v8f;

// ---------------- degree / normalization ----------------

__global__ void k_init_deg(float* __restrict__ deg) {
    int i = blockIdx.x * blockDim.x + threadIdx.x;
    if (i < N_NODES) deg[i] = 1.0f;   // self-loop contributes 1
}

__global__ void k_count_deg(const long long* __restrict__ col, float* __restrict__ deg) {
    int e = blockIdx.x * blockDim.x + threadIdx.x;
    if (e < E_EDGES) atomicAdd(&deg[(int)col[e]], 1.0f);
}

__global__ void k_dinv(const float* __restrict__ deg, float* __restrict__ dinv) {
    int i = blockIdx.x * blockDim.x + threadIdx.x;
    if (i < N_NODES) {
        float d = deg[i];
        dinv[i] = (d > 0.0f) ? rsqrtf(d) : 0.0f;
    }
}

// ---------------- fragment-layout helpers (ISA 7.12.2, wave32) ----------------
// 16-bit A 16x32: lane = hi*16 + M (hi = lane>>4); elem e<8 -> K = hi*8+e ;
//                 elem e>=8 -> K = 16 + hi*8 + (e-8)
__device__ __forceinline__ int a_frag_idx(int m /*0..15*/, int k /*0..31*/) {
    int jj = k & 15;
    int hi = jj >> 3;
    int e  = ((k < 16) ? 0 : 8) + (k & 7);
    return (hi * 16 + m) * 16 + e;          // lane*16 + elem
}
// 16-bit B 32x16: lane = ((K>>3)&1)*16 + N ; elem = (K>>4)*8 + (K&7)
__device__ __forceinline__ int b_frag_idx(int k /*0..31*/, int n /*0..15*/) {
    return (((k >> 3) & 1) * 16 + n) * 16 + ((k >> 4) * 8 + (k & 7));
}

// ---------------- fused MLP (WMMA, swizzled-fragment LDS) ----------------
// LDS layout:
//   phase 1: sAf [2 rg][32 lanes][16] f16 @0     (2048 B)
//            sBf [16 ntile][32][16]  f16 @2048   (16384 B)
//   phase 2: sHf [2 rg][8 kchunk][32][16] f16 @0 (16384 B, reuses phase-1 space)
//            sW2f [3 ntile][8 kchunk][32][16] f16 @18432 (24576 B)
#define SMEM_BYTES (18432 + 24576)

__global__ __launch_bounds__(256) void k_mlp(
    const float* __restrict__ x,  const float* __restrict__ W1,
    const float* __restrict__ b1, const float* __restrict__ W2,
    const float* __restrict__ b2, const float* __restrict__ temp,
    float* __restrict__ h0, float* __restrict__ hidden)
{
    __shared__ __align__(16) char smem[SMEM_BYTES];
    _Float16* sAf  = (_Float16*)(smem);             // 1024 halves
    _Float16* sBf  = (_Float16*)(smem + 2048);      // 8192 halves
    _Float16* sHf  = (_Float16*)(smem);             // 8192 halves (reuse)
    _Float16* sW2f = (_Float16*)(smem + 18432);     // 12288 halves

    const int tid   = threadIdx.x;
    const int lane  = tid & 31;
    const int wave  = tid >> 5;
    const int halfl = lane & 15;
    const int hi    = lane >> 4;
    const int rowBase = blockIdx.x * 32;

    // stage W2 in B-fragment layout (f32->f16, pad 40 -> 48 cols); runs once
    for (int i = tid; i < HID * 48; i += 256) {
        int k = i / 48, n = i % 48;
        float v = (n < OUT_CH) ? W2[k * OUT_CH + n] : 0.0f;
        int kchunk = k >> 5, kin = k & 31;
        int idx = ((n >> 4) * 8 + kchunk) * 512 + b_frag_idx(kin, n & 15);
        sW2f[idx] = (_Float16)v;
    }

    // ---- GEMM1: 8 waves = 2 rowgroups x 4 colgroups; wave -> 16 rows x 64 cols ----
    const int rg = wave >> 2;          // 0..1
    const int cg = wave & 3;           // 0..3
    v8f acc[4] = {};

    for (int kc = 0; kc < IN_CH; kc += 32) {
        const bool full = (kc + 32 <= IN_CH);
        __syncthreads();

        // stage A: 32 rows x 32 k -> fragment layout (one float4 per thread)
        {
            int r  = tid >> 3;              // 0..31
            int j0 = (tid & 7) * 4;         // 0..28
            const float* xp = x + (size_t)(rowBase + r) * IN_CH + kc + j0;
            float4 vx;
            if (full) {
                vx = *(const float4*)xp;
            } else {
                vx.x = (kc + j0 + 0 < IN_CH) ? xp[0] : 0.0f;
                vx.y = (kc + j0 + 1 < IN_CH) ? xp[1] : 0.0f;
                vx.z = (kc + j0 + 2 < IN_CH) ? xp[2] : 0.0f;
                vx.w = (kc + j0 + 3 < IN_CH) ? xp[3] : 0.0f;
            }
            #pragma unroll
            for (int u = 0; u < 4; ++u) {
                int j = j0 + u;
                sAf[(r >> 4) * 512 + a_frag_idx(r & 15, j)] = (_Float16)((&vx.x)[u]);
            }
        }
        // stage B: 32 k x 256 n -> fragment layout (float4 along n)
        #pragma unroll
        for (int it = 0; it < 8; ++it) {
            int i  = tid + 256 * it;        // float4 index 0..2047
            int k  = i >> 6;                // 0..31
            int n0 = (i & 63) * 4;          // 0..252
            float4 vw = {0.f, 0.f, 0.f, 0.f};
            if (full || (kc + k < IN_CH))
                vw = *(const float4*)&W1[(size_t)(kc + k) * HID + n0];
            #pragma unroll
            for (int u = 0; u < 4; ++u) {
                int n = n0 + u;
                sBf[(n >> 4) * 512 + b_frag_idx(k, n & 15)] = (_Float16)((&vw.x)[u]);
            }
        }
        __syncthreads();

        const v16h a = *(const v16h*)(sAf + rg * 512 + lane * 16);
        #pragma unroll
        for (int t = 0; t < 4; ++t) {
            const v16h b = *(const v16h*)(sBf + (cg * 4 + t) * 512 + lane * 16);
            acc[t] = __builtin_amdgcn_wmma_f32_16x16x32_f16(
                false, a, false, b, (short)0, acc[t], false, false);
        }
    }
    __syncthreads();

    // bias + relu; scatter h1 tile into sHf in GEMM2 A-fragment layout
    // C layout: elem v -> M = hi*8+v, N = halfl  (N becomes GEMM2's K)
    #pragma unroll
    for (int t = 0; t < 4; ++t) {
        int kg = cg * 64 + t * 16 + halfl;          // GEMM2 K index (0..255)
        float bias = b1[kg];
        int kchunk = kg >> 5, kin = kg & 31;
        #pragma unroll
        for (int v = 0; v < 8; ++v) {
            float val = acc[t][v] + bias;
            val = val > 0.0f ? val : 0.0f;
            int m = hi * 8 + v;                      // row within rowgroup tile
            sHf[(rg * 8 + kchunk) * 512 + a_frag_idx(m, kin)] = (_Float16)val;
        }
    }
    __syncthreads();

    // ---- GEMM2: 6 output tiles (2 rowgroups x 3 coltiles), waves 0..5 ----
    if (wave < 6) {
        const int rg2 = wave / 3;
        const int ct  = wave % 3;
        v8f c2 = {};
        #pragma unroll
        for (int kchunk = 0; kchunk < 8; ++kchunk) {
            const v16h a = *(const v16h*)(sHf  + (rg2 * 8 + kchunk) * 512 + lane * 16);
            const v16h b = *(const v16h*)(sW2f + (ct  * 8 + kchunk) * 512 + lane * 16);
            c2 = __builtin_amdgcn_wmma_f32_16x16x32_f16(
                false, a, false, b, (short)0, c2, false, false);
        }
        const int n = ct * 16 + halfl;
        if (n < OUT_CH) {
            float bias = b2[n];
            float t0 = temp[0];
            #pragma unroll
            for (int v = 0; v < 8; ++v) {
                int m = rowBase + rg2 * 16 + hi * 8 + v;
                float val = c2[v] + bias;
                h0[(size_t)m * OUT_CH + n]     = val;
                hidden[(size_t)m * OUT_CH + n] = t0 * val;
            }
        }
    }
}

// ---------------- propagation (L2-resident gather/scatter) ----------------

__global__ void k_selfloop(const float* __restrict__ h, float* __restrict__ hnew,
                           const float* __restrict__ dinv) {
    int i = blockIdx.x * blockDim.x + threadIdx.x;
    if (i < N_NODES * OUT_CH) {
        float d = dinv[i / OUT_CH];
        hnew[i] = d * d * h[i];
    }
}

// 8 lanes per edge, 5 channels per lane: contiguous 32B gathers/scatters
__global__ void k_edge_scatter(const float* __restrict__ h, float* __restrict__ hnew,
                               const long long* __restrict__ row,
                               const long long* __restrict__ col,
                               const float* __restrict__ dinv) {
    long long g = (long long)blockIdx.x * blockDim.x + threadIdx.x;
    long long e = g >> 3;
    int l8 = (int)(g & 7);
    if (e < E_EDGES) {
        int r = (int)row[e];
        int c = (int)col[e];
        float w = dinv[r] * dinv[c];
        const float* src = h + (size_t)r * OUT_CH;
        float* dst = hnew + (size_t)c * OUT_CH;
        __builtin_prefetch(src, 0, 0);   // global_prefetch_b8
        #pragma unroll
        for (int j = 0; j < 5; ++j) {
            int chv = l8 + 8 * j;
            atomicAdd(&dst[chv], w * src[chv]);
        }
    }
}

__global__ void k_axpy(float* __restrict__ hidden, const float* __restrict__ hnew,
                       const float* __restrict__ temp, int k) {
    int i = blockIdx.x * blockDim.x + threadIdx.x;
    if (i < N_NODES * OUT_CH) hidden[i] += temp[k + 1] * hnew[i];
}

// ---------------- launch ----------------

extern "C" void kernel_launch(void* const* d_in, const int* in_sizes, int n_in,
                              void* d_out, int out_size, void* d_ws, size_t ws_size,
                              hipStream_t stream) {
    const float*     x    = (const float*)d_in[0];
    const long long* ei   = (const long long*)d_in[1];   // int64 [2, E]
    const float*     W1   = (const float*)d_in[2];
    const float*     b1   = (const float*)d_in[3];
    const float*     W2   = (const float*)d_in[4];
    const float*     b2   = (const float*)d_in[5];
    const float*     temp = (const float*)d_in[6];
    float* hidden = (float*)d_out;

    const long long* row = ei;
    const long long* col = ei + E_EDGES;

    // scratch: deg | dinv | hA | hB
    float* ws   = (float*)d_ws;
    float* deg  = ws;
    float* dinv = ws + N_NODES;
    float* hA   = ws + 2 * (size_t)N_NODES;
    float* hB   = hA + (size_t)N_NODES * OUT_CH;

    const int T = 256;
    k_init_deg <<<(N_NODES + T - 1) / T, T, 0, stream>>>(deg);
    k_count_deg<<<(E_EDGES + T - 1) / T, T, 0, stream>>>(col, deg);
    k_dinv     <<<(N_NODES + T - 1) / T, T, 0, stream>>>(deg, dinv);

    // fused MLP: h0 -> hA, hidden = temp[0]*h0
    k_mlp<<<N_NODES / 32, 256, 0, stream>>>(x, W1, b1, W2, b2, temp, hA, hidden);

    const int NC = N_NODES * OUT_CH;
    float* cur = hA;
    float* nxt = hB;
    for (int k = 0; k < K_HOPS; ++k) {
        k_selfloop<<<(NC + T - 1) / T, T, 0, stream>>>(cur, nxt, dinv);
        long long nthr = (long long)E_EDGES * 8;
        k_edge_scatter<<<(unsigned)((nthr + T - 1) / T), T, 0, stream>>>(cur, nxt, row, col, dinv);
        k_axpy<<<(NC + T - 1) / T, T, 0, stream>>>(hidden, nxt, temp, k);
        float* t = cur; cur = nxt; nxt = t;
    }
}
